// DotProductWithLinear_49555332661502
// MI455X (gfx1250) — matrix operand order repaired and checked by reading, not compiled, not run on hardware
//
#include <hip/hip_runtime.h>
#include <hip/hip_bf16.h>

// ---------------- problem constants ----------------
#define BATCH     16384
#define DENSE_IN  512
#define N_SPARSE  26
#define EMB       128
#define OUT_DIM   512
#define NI        27          // 1 + N_SPARSE
#define TRI       378         // NI*(NI+1)/2
#define DOT_OUT   890         // DENSE_IN + TRI
#define KPAD      896         // DOT_OUT padded to multiple of 32
#define TROW      (NI * EMB)  // 3456 elements per sample in T
#define LDSROW    904         // padded LDS row stride (452 dwords; 452%64=4 -> no bank conflicts)

typedef __attribute__((ext_vector_type(16))) __bf16 bf16x16;
typedef __attribute__((ext_vector_type(8)))  __bf16 bf16x8;
typedef __attribute__((ext_vector_type(8)))  float  f32x8;

// D = A(16x32 bf16) * B(32x16 bf16) + C(16x16 f32)
static __device__ __forceinline__ f32x8 wmma_bf16(bf16x16 a, bf16x16 b, f32x8 c) {
  return __builtin_amdgcn_wmma_f32_16x16x32_bf16(false, a, false, b, (short)0, c,
                                                 false, false);
}

// A-fragment (16x32 bf16, row-major source): lane half h holds the row's
// elements K = h*16..h*16+15 (contiguous, matching the B fragments).
// Works for global or LDS-backed pointers (addrspace inferred -> ds_load_b128).
static __device__ __forceinline__ bf16x16 load_afrag(const __bf16* row, int h) {
  bf16x8 lo = *(const bf16x8*)(row + h * 16);
  bf16x8 hi = *(const bf16x8*)(row + h * 16 + 8);
  bf16x16 r;
#pragma unroll
  for (int e = 0; e < 8; ++e) { r[e] = lo[e]; r[8 + e] = hi[e]; }
  return r;
}

// A-fragment with on-the-fly f32 -> bf16 conversion
static __device__ __forceinline__ bf16x16 load_afrag_f32(const float* row, int h) {
  bf16x16 r;
#pragma unroll
  for (int e = 0; e < 8; ++e) r[e] = (__bf16)row[h * 16 + e];
#pragma unroll
  for (int e = 0; e < 8; ++e) r[8 + e] = (__bf16)row[h * 16 + 8 + e];
  return r;
}

static __device__ __forceinline__ bf16x16 zero_frag() {
  bf16x16 r;
#pragma unroll
  for (int e = 0; e < 16; ++e) r[e] = (__bf16)0.0f;
  return r;
}

// gfx1250 async global->LDS copy, tracked by ASYNCcnt (CDNA5 ISA).
// lds_off: byte offset within the wave's LDS allocation (low 32 bits of a
// generic shared pointer per the ISA aperture mapping).
static __device__ __forceinline__ void async_load_b128(unsigned lds_off,
                                                       const __bf16* gaddr) {
  asm volatile("global_load_async_to_lds_b128 %0, %1, off"
               :: "v"(lds_off), "v"(gaddr) : "memory");
}

static __device__ __forceinline__ void wait_asynccnt0() {
  asm volatile("s_wait_asynccnt 0x0" ::: "memory");
}

// ---------------- weight / input conversion (bf16 staging) ----------------
__global__ void k_convert_wp(const float* __restrict__ Wp, __bf16* __restrict__ wpb) {
  int i = blockIdx.x * 256 + threadIdx.x;           // EMB*DENSE_IN = 65536
  if (i < EMB * DENSE_IN) wpb[i] = (__bf16)Wp[i];
}

__global__ void k_convert_wo(const float* __restrict__ Wo, __bf16* __restrict__ wob) {
  int i = blockIdx.x * 256 + threadIdx.x;           // OUT_DIM*KPAD = 458752
  if (i >= OUT_DIM * KPAD) return;
  int n = i / KPAD, k = i % KPAD;
  wob[i] = (k < DOT_OUT) ? (__bf16)Wo[n * DOT_OUT + k] : (__bf16)0.0f;
}

__global__ void k_convert_sparse(const float* __restrict__ sp, __bf16* __restrict__ T) {
  long long i = (long long)blockIdx.x * 256 + threadIdx.x; // B*26*128
  if (i >= (long long)BATCH * N_SPARSE * EMB) return;
  long long b = i / (N_SPARSE * EMB);
  int rem = (int)(i % (N_SPARSE * EMB));
  T[b * TROW + EMB + rem] = (__bf16)sp[i];          // T rows 1..26
}

// dense -> X cols 0..511 (bf16) and zero the pad cols 890..895
__global__ void k_convert_dense(const float* __restrict__ dn, __bf16* __restrict__ X) {
  long long i = (long long)blockIdx.x * 256 + threadIdx.x; // B*(512+6)
  if (i >= (long long)BATCH * (DENSE_IN + KPAD - DOT_OUT)) return;
  long long b = i / (DENSE_IN + KPAD - DOT_OUT);
  int k = (int)(i % (DENSE_IN + KPAD - DOT_OUT));
  if (k < DENSE_IN)
    X[b * KPAD + k] = (__bf16)dn[b * DENSE_IN + k];
  else
    X[b * KPAD + DOT_OUT + (k - DENSE_IN)] = (__bf16)0.0f;
}

// ---------------- stage 1: dense projection -> T row 0 ----------------
// One wave per 16-sample tile. 8 N-tiles x 16 K-steps = 128 WMMA per wave.
__global__ void k_proj(const float* __restrict__ dense, const __bf16* __restrict__ wpb,
                       const float* __restrict__ bp, __bf16* __restrict__ T) {
  int wave = (blockIdx.x * 256 + threadIdx.x) >> 5;
  int lane = threadIdx.x & 31;
  int b0 = wave * 16;
  if (b0 >= BATCH) return;                          // uniform per wave
  int h = lane >> 4, l = lane & 15;

  const float* arow = dense + (long long)(b0 + l) * DENSE_IN;

#pragma unroll
  for (int nt = 0; nt < 8; ++nt) {
    int n0 = nt * 16;
    f32x8 acc = {};
    const __bf16* brow = wpb + (long long)(n0 + l) * DENSE_IN; // Wp row n = B col n
    for (int k0 = 0; k0 < DENSE_IN; k0 += 32) {
      bf16x16 a = load_afrag_f32(arow + k0, h);
      bf16x16 b = *(const bf16x16*)(brow + k0 + h * 16);       // 32B aligned
      acc = wmma_bf16(a, b, acc);
    }
    float bias = bp[n0 + l];
#pragma unroll
    for (int r = 0; r < 8; ++r) {                   // C: row m = r + 8h, col n0+l
      int m = r + 8 * h;
      T[(long long)(b0 + m) * TROW + (n0 + l)] = (__bf16)(acc[r] + bias);
    }
  }
}

// ---------------- stage 2: per-sample Z = T*T^T, scatter tril ----------------
// One wave per sample; T padded 27->32 rows; tile (m0,n1) is strictly upper
// triangular and skipped -> 3 C-tiles x 4 K-steps = 12 WMMA per sample.
__global__ void k_interact(const __bf16* __restrict__ T, __bf16* __restrict__ X) {
  int wave = (blockIdx.x * 256 + threadIdx.x) >> 5;
  int lane = threadIdx.x & 31;
  if (wave >= BATCH) return;                        // uniform per wave
  int h = lane >> 4, l = lane & 15;

  const __bf16* Tb = T + (long long)wave * TROW;
  __bf16* Xb = X + (long long)wave * KPAD + DENSE_IN;

  f32x8 C00 = {}, C10 = {}, C11 = {};
  int r1 = 16 + l;                                  // row index for m/n-tile 1

  for (int k0 = 0; k0 < EMB; k0 += 32) {
    bf16x16 a0 = load_afrag(Tb + (long long)l * EMB + k0, h);
    bf16x16 a1 = (r1 < NI) ? load_afrag(Tb + (long long)r1 * EMB + k0, h) : zero_frag();
    // B = T^T : B[K][N] = T[N][K]; lane's column n reads T row n contiguously
    bf16x16 b0 = *(const bf16x16*)(Tb + (long long)l * EMB + k0 + h * 16);
    bf16x16 b1 = (r1 < NI) ? *(const bf16x16*)(Tb + (long long)r1 * EMB + k0 + h * 16)
                           : zero_frag();
    C00 = wmma_bf16(a0, b0, C00);
    C10 = wmma_bf16(a1, b0, C10);
    C11 = wmma_bf16(a1, b1, C11);
  }

  // scatter lower-triangular entries: X[b, 512 + i*(i+1)/2 + j] for j <= i
#pragma unroll
  for (int r = 0; r < 8; ++r) {
    {                                               // tile (0,0): i = r+8h, j = l
      int i = r + 8 * h, j = l;
      if (j <= i) Xb[i * (i + 1) / 2 + j] = (__bf16)C00[r];
    }
    {                                               // tile (1,0): i = 16+r+8h, j = l
      int i = 16 + r + 8 * h, j = l;
      if (i < NI) Xb[i * (i + 1) / 2 + j] = (__bf16)C10[r];
    }
    {                                               // tile (1,1): i = 16+r+8h, j = 16+l
      int i = 16 + r + 8 * h, j = 16 + l;
      if (i < NI && j <= i) Xb[i * (i + 1) / 2 + j] = (__bf16)C11[r];
    }
  }
}

// ---------------- stage 3: out = X(896) @ WoT + bo ----------------
// Block = 32 rows; 8 waves each own 64 output cols (32x64 per wave).
// A-panel (32 x 896 bf16, 56 KB) staged once in LDS via async b128 copies;
// A-fragments come from LDS (ds_load_b128, conflict-free via padded stride),
// B-fragments stream from L2-resident bf16 Wo. 28 K-steps x 8 WMMA = 224/wave.
__global__ void k_outgemm(const __bf16* __restrict__ X, const __bf16* __restrict__ wob,
                          const float* __restrict__ bo, float* __restrict__ out) {
  __shared__ __bf16 sA[32 * LDSROW];                // 57,856 bytes

  int b0 = blockIdx.x * 32;
  int tid = threadIdx.x;
  int w = tid >> 5;
  int lane = tid & 31;
  int h = lane >> 4, l = lane & 15;
  int n0 = w * 64;

  // cooperative async stage of the 32x896 A-panel (112 x 16B chunks per row)
  {
    unsigned sbase = (unsigned)(uintptr_t)(&sA[0]);
    const __bf16* Xbase = X + (long long)b0 * KPAD;
#pragma unroll
    for (int c = tid; c < 32 * 112; c += 256) {
      int row = c / 112, cc = c % 112;
      async_load_b128(sbase + (unsigned)(row * LDSROW + cc * 8) * 2u,
                      Xbase + (long long)row * KPAD + cc * 8);
    }
    wait_asynccnt0();
    __syncthreads();
  }

  f32x8 acc[2][4] = {{{}, {}, {}, {}}, {{}, {}, {}, {}}};

  for (int k0 = 0; k0 < KPAD; k0 += 32) {
    bf16x16 a0 = load_afrag(&sA[(0 + l) * LDSROW + k0], h);   // rows 0..15
    bf16x16 a1 = load_afrag(&sA[(16 + l) * LDSROW + k0], h);  // rows 16..31
#pragma unroll
    for (int t = 0; t < 4; ++t) {
      bf16x16 q = *(const bf16x16*)(wob + (long long)(n0 + t * 16 + l) * KPAD +
                                    k0 + h * 16);
      acc[0][t] = wmma_bf16(a0, q, acc[0][t]);
      acc[1][t] = wmma_bf16(a1, q, acc[1][t]);
    }
  }

#pragma unroll
  for (int mt = 0; mt < 2; ++mt) {
#pragma unroll
    for (int t = 0; t < 4; ++t) {
      int n = n0 + t * 16 + l;
      float bias = bo[n];
#pragma unroll
      for (int r = 0; r < 8; ++r) {
        int m = mt * 16 + r + 8 * h;
        out[(long long)(b0 + m) * OUT_DIM + n] = acc[mt][t][r] + bias;
      }
    }
  }
}

// ---------------- launch ----------------
extern "C" void kernel_launch(void* const* d_in, const int* in_sizes, int n_in,
                              void* d_out, int out_size, void* d_ws, size_t ws_size,
                              hipStream_t stream) {
  const float* dense  = (const float*)d_in[0];
  const float* sparse = (const float*)d_in[1];
  const float* Wp     = (const float*)d_in[2];
  const float* bp     = (const float*)d_in[3];
  const float* Wo     = (const float*)d_in[4];
  const float* bo     = (const float*)d_in[5];
  float* out = (float*)d_out;

  const size_t T_BYTES  = (size_t)BATCH * TROW * sizeof(__bf16);   // 113,246,208
  const size_t X_BYTES  = (size_t)BATCH * KPAD * sizeof(__bf16);   //  29,360,128
  const size_t WP_BYTES = (size_t)EMB * DENSE_IN * sizeof(__bf16); //     131,072

  char* ws = (char*)d_ws;
  __bf16* Tbuf = (__bf16*)(ws);
  __bf16* Xbuf = (__bf16*)(ws + T_BYTES);
  __bf16* WpB  = (__bf16*)(ws + T_BYTES + X_BYTES);
  __bf16* WoB  = (__bf16*)(ws + T_BYTES + X_BYTES + WP_BYTES);

  // bf16 staging: T (113 MB) + X (29 MB) + weights (<1 MB) fit in the 192 MB L2
  k_convert_wp    <<<(EMB * DENSE_IN + 255) / 256, 256, 0, stream>>>(Wp, WpB);
  k_convert_wo    <<<(OUT_DIM * KPAD + 255) / 256, 256, 0, stream>>>(Wo, WoB);
  k_convert_sparse<<<(BATCH * N_SPARSE * EMB + 255) / 256, 256, 0, stream>>>(sparse, Tbuf);
  k_convert_dense <<<(BATCH * (DENSE_IN + KPAD - DOT_OUT) + 255) / 256, 256, 0,
                    stream>>>(dense, Xbuf);

  k_proj    <<<BATCH / 16 / 8, 256, 0, stream>>>(dense, WpB, bp, Tbuf);
  k_interact<<<BATCH / 8,      256, 0, stream>>>(Tbuf, Xbuf);
  k_outgemm <<<BATCH / 32,     256, 0, stream>>>(Xbuf, WoB, bo, out);
}
